// SelfAttention_65377992179726
// MI455X (gfx1250) — matrix operand order repaired and checked by reading, not compiled, not run on hardware
//
#include <hip/hip_runtime.h>
#include <hip/hip_bf16.h>
#include <math.h>

typedef __bf16 bf16_t;
typedef __attribute__((ext_vector_type(16))) __bf16 v16bf;
typedef __attribute__((ext_vector_type(8)))  float  v8f;
typedef __attribute__((ext_vector_type(4)))  unsigned int v4u32;
typedef __attribute__((ext_vector_type(8)))  int v8i32;
typedef __attribute__((ext_vector_type(4)))  int v4i32;

#define RR 256
#define CC 384
#define EE 384
#define HH 12
#define DD 32
#define NEGF (-3.4028234663852886e38f)

// ---------------------------------------------------------------------------
// WMMA fragment helpers (wave32; A and B share a contiguous k order, C/D per ISA)
// A 16x32 bf16: element e (v=e/2, sub=e&1): K = 16*half + 2v + sub
//   -> per lane: one contiguous 16-element run => 2x global_load_b128
// B 32x16 bf16: N = lane&15 ; K = 16*half + 2v + sub
// C/D 16x16 f32: N = lane&15 ; M = v + 8*half
// ---------------------------------------------------------------------------

__device__ __forceinline__ v16bf load_a_frag(const bf16_t* __restrict__ base, int ld) {
    const int lane = threadIdx.x & 31;
    const int m = lane & 15, half = lane >> 4;
    v16bf a{};
#pragma unroll
    for (int e = 0; e < 16; ++e) {
        const int v = e >> 1, sub = e & 1;
        const int k = 16 * half + 2 * v + sub;
        a[e] = base[(size_t)m * ld + k];
    }
    return a;
}

__device__ __forceinline__ v16bf load_bT_frag(const bf16_t* __restrict__ base, int ld) {
    // B element (k,n) sourced from row-major transposed storage: base[n*ld + k]
    // per lane: 16 contiguous bf16 => 2x b128 loads (global or ds)
    const int lane = threadIdx.x & 31;
    const int n = lane & 15, half = lane >> 4;
    v16bf b{};
#pragma unroll
    for (int e = 0; e < 16; ++e) {
        const int v = e >> 1, sub = e & 1;
        const int k = 16 * half + 2 * v + sub;
        b[e] = base[(size_t)n * ld + k];
    }
    return b;
}

__device__ __forceinline__ void store_d_f32(float* __restrict__ base, int ld, v8f d) {
    const int lane = threadIdx.x & 31;
    const int n = lane & 15, half = lane >> 4;
#pragma unroll
    for (int v = 0; v < 8; ++v)
        base[(size_t)(v + 8 * half) * ld + n] = d[v];
}

__device__ __forceinline__ void store_d_bf16(bf16_t* __restrict__ base, int ld, v8f d) {
    const int lane = threadIdx.x & 31;
    const int n = lane & 15, half = lane >> 4;
#pragma unroll
    for (int v = 0; v < 8; ++v)
        base[(size_t)(v + 8 * half) * ld + n] = (bf16_t)d[v];
}

__device__ __forceinline__ v8f wmma_bf16(v16bf a, v16bf b, v8f c) {
    return __builtin_amdgcn_wmma_f32_16x16x32_bf16(false, a, false, b, (short)0, c,
                                                   false, false);
}

// ---------------------------------------------------------------------------
// K0: f32 -> bf16 conversion of x
// ---------------------------------------------------------------------------
__global__ void cvt_bf16_kernel(const float* __restrict__ x, bf16_t* __restrict__ xb,
                                size_t n) {
    for (size_t i = (size_t)blockIdx.x * blockDim.x + threadIdx.x; i < n;
         i += (size_t)gridDim.x * blockDim.x)
        xb[i] = (bf16_t)x[i];
}

// ---------------------------------------------------------------------------
// K0b: pack W (f32, [k][n]) -> WT (bf16, [n][k]) so B fragments load as 2xb128
// ---------------------------------------------------------------------------
__global__ void wt_pack_kernel(const float* __restrict__ W, bf16_t* __restrict__ WT) {
    const int i = blockIdx.x * blockDim.x + threadIdx.x;
    if (i < EE * EE) {
        const int n = i / EE, k = i % EE;
        WT[i] = (bf16_t)W[(size_t)k * EE + n];
    }
}

// ---------------------------------------------------------------------------
// K1: Y(98304x384) = Xb @ W. Block = 8 waves; all waves share one 32-column
// slice of WT, staged into LDS by the Tensor Data Mover (tensor_load_to_lds,
// s_wait_tensorcnt). Each wave computes a 2x2 grid of 16x16 tiles.
// ---------------------------------------------------------------------------
__global__ void proj_gemm_kernel(const bf16_t* __restrict__ Xb,
                                 const bf16_t* __restrict__ WT,
                                 bf16_t* __restrict__ Y) {
    __shared__ __align__(16) bf16_t ldsW[32 * EE];  // 24 KB: 32 n-rows x 384 k

    const int NPAIRS = EE / 32;                     // 12
    const int npair = blockIdx.x % NPAIRS;
    const int mblk  = blockIdx.x / NPAIRS;          // 0..383, 16 m-tiles each

    if (threadIdx.x < 32) {
        // --- TDM descriptor (CDNA5 ISA ch.8): 2D bf16 tensor 384x384,
        //     tile = rows [npair*32, +32) x 384 cols -> LDS
        const unsigned lds_off = (unsigned)(uintptr_t)ldsW;
        const unsigned long long ga =
            (unsigned long long)(uintptr_t)(WT + (size_t)npair * 32 * EE);
        v4u32 g0 = {};
        g0[0] = 1u;                                  // count=1 valid descriptor
        g0[1] = lds_off;                             // lds_addr
        g0[2] = (unsigned)(ga & 0xffffffffu);        // global_addr[31:0]
        g0[3] = (unsigned)((ga >> 32) & 0x1ffffffu) | (2u << 30);  // addr hi | type=2
        v8i32 g1 = {};
        g1[0] = 0x10000;                 // workgroup_mask=0, data_size=1 (2B)
        g1[1] = (EE & 0xffff) << 16;     // tensor_dim0[15:0] @ bits63:48
        g1[2] = (EE & 0xffff) << 16;     // tensor_dim1[15:0] @ bits95:80
        g1[3] = (EE & 0xffff) << 16;     // tile_dim0=384 @ bits127:112
        g1[4] = 32;                      // tile_dim1=32
        g1[5] = EE;                      // tensor_dim0_stride low32
        v4i32 g2 = {}, g3 = {};
        v8i32 g4 = {};
        __builtin_amdgcn_tensor_load_to_lds(g0, g1, g2, g3, g4, 0);
        __builtin_amdgcn_s_wait_tensorcnt((short)0);
    }
    __syncthreads();

    const int wave = threadIdx.x >> 5;               // 0..7
    const int mt0 = mblk * 16 + wave * 2;            // two m-tiles per wave
    const bf16_t* arow0 = Xb + (size_t)mt0 * 16 * EE;
    const bf16_t* arow1 = arow0 + (size_t)16 * EE;

    v8f acc00 = {}, acc01 = {}, acc10 = {}, acc11 = {};
#pragma unroll 2
    for (int kk = 0; kk < EE; kk += 32) {
        v16bf a0 = load_a_frag(arow0 + kk, EE);
        v16bf a1 = load_a_frag(arow1 + kk, EE);
        v16bf b0 = load_bT_frag(ldsW + kk, EE);              // ds_load_b128 x2
        v16bf b1 = load_bT_frag(ldsW + 16 * EE + kk, EE);
        acc00 = wmma_bf16(a0, b0, acc00);
        acc01 = wmma_bf16(a0, b1, acc01);
        acc10 = wmma_bf16(a1, b0, acc10);
        acc11 = wmma_bf16(a1, b1, acc11);
    }
    const int nt0 = npair * 2;
    bf16_t* yb = Y + (size_t)mt0 * 16 * EE + nt0 * 16;
    store_d_bf16(yb, EE, acc00);
    store_d_bf16(yb + 16, EE, acc01);
    store_d_bf16(yb + (size_t)16 * EE, EE, acc10);
    store_d_bf16(yb + (size_t)16 * EE + 16, EE, acc11);
}

// ---------------------------------------------------------------------------
// K2: gating softmax sw[r,c,h]. One wave per (c,h); lane == d.
// ---------------------------------------------------------------------------
__global__ void sw_kernel(const float* __restrict__ x, const float* __restrict__ Wq_s,
                          const bf16_t* __restrict__ Ksp,
                          const int* __restrict__ seq_mask, float* __restrict__ sw) {
    const int wave = (int)((blockIdx.x * blockDim.x + threadIdx.x) >> 5);
    const int lane = threadIdx.x & 31;
    const int c = wave / HH, h = wave % HH;
    if (c >= CC) return;

    const float* q0 = x + (size_t)c * EE;  // x[b=0, r=0, c, :]
    float qs = 0.f;
    for (int e = 0; e < EE; ++e)
        qs += q0[e] * Wq_s[(size_t)e * EE + h * DD + lane];
    qs *= 0.17677669529663687f;  // 1/sqrt(32)

    float logit[RR / 32];
#pragma unroll 1
    for (int r = 0; r < RR; ++r) {
        float v = qs * (float)Ksp[((size_t)r * CC + c) * EE + h * DD + lane];
#pragma unroll
        for (int off = 16; off; off >>= 1) v += __shfl_xor(v, off, 32);
        if ((r & 31) == lane) logit[r >> 5] = v;
    }
#pragma unroll
    for (int t = 0; t < RR / 32; ++t)
        if (seq_mask[t * 32 + lane] == 0) logit[t] = NEGF;

    float mx = logit[0];
#pragma unroll
    for (int t = 1; t < RR / 32; ++t) mx = fmaxf(mx, logit[t]);
#pragma unroll
    for (int off = 16; off; off >>= 1) mx = fmaxf(mx, __shfl_xor(mx, off, 32));
    float sum = 0.f;
#pragma unroll
    for (int t = 0; t < RR / 32; ++t) {
        logit[t] = __expf(logit[t] - mx);
        sum += logit[t];
    }
#pragma unroll
    for (int off = 16; off; off >>= 1) sum += __shfl_xor(sum, off, 32);
    const float inv = 1.f / sum;
#pragma unroll
    for (int t = 0; t < RR / 32; ++t) {
        const int r = t * 32 + lane;
        sw[((size_t)r * CC + c) * HH + h] = logit[t] * inv;
    }
}

// ---------------------------------------------------------------------------
// K2b: Vt[r,h,d,c] = Vp[r,c,h,d]  (d-major -> c-major for out-GEMM B frags)
// ---------------------------------------------------------------------------
__global__ void vt_pack_kernel(const bf16_t* __restrict__ Vp, bf16_t* __restrict__ Vt) {
    const size_t n = (size_t)RR * CC * EE;
    for (size_t i = (size_t)blockIdx.x * blockDim.x + threadIdx.x; i < n;
         i += (size_t)gridDim.x * blockDim.x) {
        const size_t rc = i / EE;
        const int e = (int)(i % EE);
        const int r = (int)(rc / CC), c = (int)(rc % CC);
        const int h = e / DD, d = e % DD;
        Vt[(((size_t)r * HH + h) * DD + d) * CC + c] = Vp[i];
    }
}

// ---------------------------------------------------------------------------
// K3: Q *= SCALING/sqrt(C) * sw[r,c,h]  (in place, bf16)
// ---------------------------------------------------------------------------
__global__ void scale_q_kernel(bf16_t* __restrict__ Qp, const float* __restrict__ sw) {
    const size_t n = (size_t)RR * CC * EE;
    const float scl = 0.17677669529663687f / sqrtf((float)CC);
    for (size_t i = (size_t)blockIdx.x * blockDim.x + threadIdx.x; i < n;
         i += (size_t)gridDim.x * blockDim.x) {
        const size_t rc = i / EE;
        const int e = (int)(i % EE);
        const int h = e / DD;
        Qp[i] = (bf16_t)((float)Qp[i] * scl * sw[rc * HH + h]);
    }
}

// ---------------------------------------------------------------------------
// K4: QK[h,i,j] = sum_{r,d} Qw[r,i,h,d]*Kp[r,j,h,d]. 2x2 tiles per wave,
// K loop over r: 4 fragment loads -> 4 wmma per step, 256 steps.
// ---------------------------------------------------------------------------
__global__ void qk_gemm_kernel(const bf16_t* __restrict__ Qw,
                               const bf16_t* __restrict__ Kp,
                               float* __restrict__ QK) {
    const int TP = CC / 32;  // 12 tile-pairs
    const int wave = (int)((blockIdx.x * blockDim.x + threadIdx.x) >> 5);
    const int h = wave / (TP * TP);
    if (h >= HH) return;
    const int rem = wave % (TP * TP);
    const int itp = rem / TP, jtp = rem % TP;

    const bf16_t* Ab = Qw + (size_t)itp * 32 * EE + h * DD;
    const bf16_t* Bb = Kp + (size_t)jtp * 32 * EE + h * DD;
    v8f acc00 = {}, acc01 = {}, acc10 = {}, acc11 = {};
#pragma unroll 2
    for (int r = 0; r < RR; ++r) {
        const size_t roff = (size_t)r * CC * EE;
        __builtin_prefetch(Ab + roff + (size_t)CC * EE, 0, 1);  // global_prefetch_b8
        __builtin_prefetch(Bb + roff + (size_t)CC * EE, 0, 1);
        v16bf a0 = load_a_frag(Ab + roff, EE);
        v16bf a1 = load_a_frag(Ab + roff + (size_t)16 * EE, EE);
        v16bf b0 = load_bT_frag(Bb + roff, EE);
        v16bf b1 = load_bT_frag(Bb + roff + (size_t)16 * EE, EE);
        acc00 = wmma_bf16(a0, b0, acc00);
        acc01 = wmma_bf16(a0, b1, acc01);
        acc10 = wmma_bf16(a1, b0, acc10);
        acc11 = wmma_bf16(a1, b1, acc11);
    }
    float* qb = QK + ((size_t)h * CC + itp * 32) * CC + jtp * 32;
    store_d_f32(qb, CC, acc00);
    store_d_f32(qb + 16, CC, acc01);
    store_d_f32(qb + (size_t)16 * CC, CC, acc10);
    store_d_f32(qb + (size_t)16 * CC + 16, CC, acc11);
}

// ---------------------------------------------------------------------------
// K5: attn = softmax_j( mask(QK) ), bf16 output. One wave per (h,i) row.
// ---------------------------------------------------------------------------
__global__ void attn_softmax_kernel(const float* __restrict__ QK,
                                    const int* __restrict__ res_mask,
                                    bf16_t* __restrict__ attn) {
    const int wave = (int)((blockIdx.x * blockDim.x + threadIdx.x) >> 5);
    const int lane = threadIdx.x & 31;
    const int h = wave / CC, i = wave % CC;
    if (h >= HH) return;
    const float* row = QK + ((size_t)h * CC + i) * CC;
    const int rmi = res_mask[i];

    float v[CC / 32];
#pragma unroll
    for (int t = 0; t < CC / 32; ++t) {
        const int j = t * 32 + lane;
        float q = row[j];
        if (rmi == 0 || res_mask[j] == 0) q = NEGF;
        v[t] = q;
    }
    float mx = v[0];
#pragma unroll
    for (int t = 1; t < CC / 32; ++t) mx = fmaxf(mx, v[t]);
#pragma unroll
    for (int off = 16; off; off >>= 1) mx = fmaxf(mx, __shfl_xor(mx, off, 32));
    float sum = 0.f;
#pragma unroll
    for (int t = 0; t < CC / 32; ++t) {
        v[t] = __expf(v[t] - mx);
        sum += v[t];
    }
#pragma unroll
    for (int off = 16; off; off >>= 1) sum += __shfl_xor(sum, off, 32);
    const float inv = 1.f / sum;
    bf16_t* orow = attn + ((size_t)h * CC + i) * CC;
#pragma unroll
    for (int t = 0; t < CC / 32; ++t)
        orow[t * 32 + lane] = (bf16_t)(v[t] * inv);
}

// ---------------------------------------------------------------------------
// K6: out[r,i,h,d] = sum_j attn[h,i,j]*V[r,j,h,d]. 2 i-tiles x 2 d-tiles per
// wave; B fragments read from packed Vt[r,h,d,c] (contiguous b128 loads).
// ---------------------------------------------------------------------------
__global__ void out_gemm_kernel(const bf16_t* __restrict__ attn,
                                const bf16_t* __restrict__ Vt,
                                float* __restrict__ out) {
    const int ITP = CC / 32;  // 12 i-tile-pairs
    const int wave = (int)((blockIdx.x * blockDim.x + threadIdx.x) >> 5);
    const int itp = wave % ITP;
    int tmp = wave / ITP;
    const int h = tmp % HH;
    const int r = tmp / HH;
    if (r >= RR) return;

    const bf16_t* Ab = attn + ((size_t)h * CC + itp * 32) * CC;
    const bf16_t* Bb = Vt + ((size_t)r * HH + h) * DD * CC;  // [d][c], ld=CC
    v8f acc00 = {}, acc01 = {}, acc10 = {}, acc11 = {};
#pragma unroll 2
    for (int jj = 0; jj < CC; jj += 32) {
        __builtin_prefetch(Bb + jj + 256, 0, 1);
        v16bf a0 = load_a_frag(Ab + jj, CC);
        v16bf a1 = load_a_frag(Ab + (size_t)16 * CC + jj, CC);
        v16bf b0 = load_bT_frag(Bb + jj, CC);                       // d-tile 0
        v16bf b1 = load_bT_frag(Bb + (size_t)16 * CC + jj, CC);     // d-tile 1
        acc00 = wmma_bf16(a0, b0, acc00);
        acc01 = wmma_bf16(a0, b1, acc01);
        acc10 = wmma_bf16(a1, b0, acc10);
        acc11 = wmma_bf16(a1, b1, acc11);
    }
    float* ob = out + ((size_t)r * CC + itp * 32) * EE + h * DD;
    store_d_f32(ob, EE, acc00);
    store_d_f32(ob + 16, EE, acc01);
    store_d_f32(ob + (size_t)16 * EE, EE, acc10);
    store_d_f32(ob + (size_t)16 * EE + 16, EE, acc11);
}

// ---------------------------------------------------------------------------
extern "C" void kernel_launch(void* const* d_in, const int* in_sizes, int n_in,
                              void* d_out, int out_size, void* d_ws, size_t ws_size,
                              hipStream_t stream) {
    const float* x    = (const float*)d_in[0];
    const float* Wq   = (const float*)d_in[1];
    const float* Wk   = (const float*)d_in[2];
    const float* Wv   = (const float*)d_in[3];
    const float* Wq_s = (const float*)d_in[4];
    const float* Wk_s = (const float*)d_in[5];
    const int* res_mask = (const int*)d_in[6];
    const int* seq_mask = (const int*)d_in[7];
    float* out = (float*)d_out;

    const size_t nx = (size_t)RR * CC * EE;  // 37,748,736 elements
    const size_t nw = (size_t)EE * EE;       // 147,456 weights

    // workspace carve-out (~396 MB)
    char* ws = (char*)d_ws;
    bf16_t* xb  = (bf16_t*)ws; ws += nx * sizeof(bf16_t);
    bf16_t* Qp  = (bf16_t*)ws; ws += nx * sizeof(bf16_t);
    bf16_t* Kp  = (bf16_t*)ws; ws += nx * sizeof(bf16_t);
    bf16_t* Vp  = (bf16_t*)ws; ws += nx * sizeof(bf16_t);
    bf16_t* Ksp = (bf16_t*)ws; ws += nx * sizeof(bf16_t);  // reused as Vt later
    float*  sw  = (float*)ws;  ws += (size_t)RR * CC * HH * sizeof(float);
    float*  QK  = (float*)ws;  ws += (size_t)HH * CC * CC * sizeof(float);
    bf16_t* attn = (bf16_t*)ws; ws += (size_t)HH * CC * CC * sizeof(bf16_t);
    bf16_t* WTq = (bf16_t*)ws; ws += nw * sizeof(bf16_t);
    bf16_t* WTk = (bf16_t*)ws; ws += nw * sizeof(bf16_t);
    bf16_t* WTv = (bf16_t*)ws; ws += nw * sizeof(bf16_t);
    bf16_t* WTks = (bf16_t*)ws;

    // K0: x -> bf16 ; pack transposed bf16 weights
    cvt_bf16_kernel<<<4096, 256, 0, stream>>>(x, xb, nx);
    wt_pack_kernel<<<(int)(nw / 256), 256, 0, stream>>>(Wq, WTq);
    wt_pack_kernel<<<(int)(nw / 256), 256, 0, stream>>>(Wk, WTk);
    wt_pack_kernel<<<(int)(nw / 256), 256, 0, stream>>>(Wv, WTv);
    wt_pack_kernel<<<(int)(nw / 256), 256, 0, stream>>>(Wk_s, WTks);

    // K1: projections. grid = 384 m-blocks x 12 n-pairs = 4608 blocks
    const int projBlocks = ((RR * CC) / 16 / 16) * (EE / 32);  // 4608
    proj_gemm_kernel<<<projBlocks, 256, 0, stream>>>(xb, WTq, Qp);
    proj_gemm_kernel<<<projBlocks, 256, 0, stream>>>(xb, WTk, Kp);
    proj_gemm_kernel<<<projBlocks, 256, 0, stream>>>(xb, WTv, Vp);
    proj_gemm_kernel<<<projBlocks, 256, 0, stream>>>(xb, WTks, Ksp);

    // K2: gating softmax (4608 waves -> 1152 blocks of 128)
    sw_kernel<<<(CC * HH * 32) / 128, 128, 0, stream>>>(x, Wq_s, Ksp, seq_mask, sw);

    // K2b: V -> Vt (reuses Ksp buffer; Ksp dead after sw_kernel)
    bf16_t* Vt = Ksp;
    vt_pack_kernel<<<4096, 256, 0, stream>>>(Vp, Vt);

    // K3: Q *= scaling * sw (in place)
    scale_q_kernel<<<4096, 256, 0, stream>>>(Qp, sw);

    // K4: per-head QK (12*12*12 = 1728 waves -> 216 blocks)
    qk_gemm_kernel<<<(HH * (CC / 32) * (CC / 32) * 32) / 256, 256, 0, stream>>>(Qp, Kp,
                                                                                QK);

    // K5: masked softmax over j (4608 waves -> 576 blocks)
    attn_softmax_kernel<<<(HH * CC * 32) / 256, 256, 0, stream>>>(QK, res_mask, attn);

    // K6: out = attn @ V (256*12*12 = 36864 waves -> 4608 blocks)
    out_gemm_kernel<<<(RR * HH * (CC / 32) * 32) / 256, 256, 0, stream>>>(attn, Vt, out);
}
